// RGAT_6004364280400
// MI455X (gfx1250) — compile-verified, matrix operand order checked by reading, not run on hardware
//
#include <hip/hip_runtime.h>
#include <hip/hip_bf16.h>

// ---------------- problem constants (match reference) ----------------
#define NN 100000     // nodes
#define EE 500000     // edges per etype
#define ETY 3         // edge types
#define LL 2          // layers
#define FF 256        // in feats == h feats
#define HH 4          // heads
#define DD 64         // head dim
#define HD 256        // H*D
#define CC 349        // classes
#define NEG_SLOPE 0.2f

typedef __attribute__((ext_vector_type(16))) __bf16 v16bf;
typedef __attribute__((ext_vector_type(8)))  float  v8f;

// ---------------- small utility kernels ----------------
__global__ void fill_f32(float* __restrict__ p, float v, long n) {
    long i = (long)blockIdx.x * blockDim.x + threadIdx.x;
    long stride = (long)gridDim.x * blockDim.x;
    for (; i < n; i += stride) p[i] = v;
}

__global__ void f32_to_bf16(const float* __restrict__ in, __bf16* __restrict__ out, long n) {
    long i = (long)blockIdx.x * blockDim.x + threadIdx.x;
    long stride = (long)gridDim.x * blockDim.x;
    for (; i < n; i += stride) out[i] = (__bf16)in[i];
}

// convert K x Nc f32 weight into Nc x K bf16 (transposed), so GEMM B-fragment
// loads become contiguous 16B runs (global_load_b128) instead of strided b16.
__global__ void f32_to_bf16_T(const float* __restrict__ in, __bf16* __restrict__ out,
                              int K, int Ncc) {
    int i = blockIdx.x * blockDim.x + threadIdx.x;   // over K*Ncc
    if (i >= K * Ncc) return;
    const int k = i / Ncc, n = i % Ncc;
    out[(size_t)n * K + k] = (__bf16)in[i];
}

// ---------------- WMMA GEMM: C[M x Nc] = A(bf16, M x 256) * B (+bias) ----------------
// B supplied TRANSPOSED: Bt is Nc x 256 (row n = output column n).
// One wave computes a 16x64 strip of C: one shared A fragment, 4 B fragments,
// 4 accumulators -> 4 v_wmma_f32_16x16x32_bf16 per 32-wide k-step.
__global__ void gemm_bf16_wmma(const __bf16* __restrict__ A,
                               const __bf16* __restrict__ Bt,
                               const float*  __restrict__ bias,   // Nc entries or nullptr
                               float*        __restrict__ C,
                               int M, int Nc) {
    const int nT64   = (Nc + 63) >> 6;                 // 64-col strips
    const int mTiles = (M + 15) >> 4;
    const int totalTiles = mTiles * nT64;

    int wave = (blockIdx.x * (int)blockDim.x + (int)threadIdx.x) >> 5;
    if (wave >= totalTiles) return;
    const int tileM = wave / nT64;
    const int tn    = wave % nT64;

    const int lane = threadIdx.x & 31;
    const int half = lane >> 4;        // 0: lanes 0-15, 1: lanes 16-31
    const int lm   = lane & 15;

    const int row  = tileM * 16 + lm;
    const int arow = (row < M) ? row : 0;   // clamped rows feed only unstored output rows
    const __bf16* __restrict__ ap = A + (size_t)arow * FF;

    int col[4];
    const __bf16* __restrict__ bp[4];
#pragma unroll
    for (int j = 0; j < 4; ++j) {
        col[j] = tn * 64 + j * 16 + lm;
        const int bc = (col[j] < Nc) ? col[j] : (Nc - 1);
        bp[j] = Bt + (size_t)bc * FF;
    }

    v8f acc0 = {}, acc1 = {}, acc2 = {}, acc3 = {};
#pragma unroll 1
    for (int k0 = 0; k0 < FF; k0 += 32) {
        const int kb = k0 + (half ? 8 : 0);
        v16bf a;
#pragma unroll
        for (int i = 0; i < 8; ++i) {          // two contiguous 16B runs
            a[i]     = ap[kb + i];
            a[8 + i] = ap[kb + 16 + i];
        }
        v16bf b0, b1, b2, b3;
#pragma unroll
        for (int i = 0; i < 8; ++i) {          // same contiguous pattern on Bt rows
            b0[i] = bp[0][kb + i];  b0[8 + i] = bp[0][kb + 16 + i];
            b1[i] = bp[1][kb + i];  b1[8 + i] = bp[1][kb + 16 + i];
            b2[i] = bp[2][kb + i];  b2[8 + i] = bp[2][kb + 16 + i];
            b3[i] = bp[3][kb + i];  b3[8 + i] = bp[3][kb + 16 + i];
        }
        acc0 = __builtin_amdgcn_wmma_f32_16x16x32_bf16(false, a, false, b0, (short)0, acc0, false, false);
        acc1 = __builtin_amdgcn_wmma_f32_16x16x32_bf16(false, a, false, b1, (short)0, acc1, false, false);
        acc2 = __builtin_amdgcn_wmma_f32_16x16x32_bf16(false, a, false, b2, (short)0, acc2, false, false);
        acc3 = __builtin_amdgcn_wmma_f32_16x16x32_bf16(false, a, false, b3, (short)0, acc3, false, false);
    }

    // C/D layout: VGPR j holds M = tileM*16 + j + 8*half, N = col.
    v8f* accs[4] = { &acc0, &acc1, &acc2, &acc3 };
#pragma unroll
    for (int j = 0; j < 4; ++j) {
        if (col[j] >= Nc) continue;
        const float bi = bias ? bias[col[j]] : 0.0f;
        const v8f& ac = *accs[j];
#pragma unroll
        for (int q = 0; q < 8; ++q) {
            const int r = tileM * 16 + half * 8 + q;
            if (r < M) C[(size_t)r * Nc + col[j]] = ac[q] + bi;
        }
    }
}

// ---------------- attention scalar kernels ----------------
// el[n,h] = feat[n,h,:] . attn_l[h,:]   er[n,h] = feat[n,h,:] . attn_r[h,:]
__global__ void el_er_kernel(const float* __restrict__ feat,
                             const float* __restrict__ al,
                             const float* __restrict__ ar,
                             float* __restrict__ el, float* __restrict__ er) {
    int i = blockIdx.x * blockDim.x + threadIdx.x;   // over N*H
    if (i >= NN * HH) return;
    const int n = i / HH, h = i % HH;
    const float* f = feat + (size_t)n * HD + h * DD;
    const float* a = al + h * DD;
    const float* b = ar + h * DD;
    float sl = 0.f, sr = 0.f;
#pragma unroll 4
    for (int d = 0; d < DD; ++d) { sl += f[d] * a[d]; sr += f[d] * b[d]; }
    el[i] = sl; er[i] = sr;
}

__device__ __forceinline__ float lrelu(float x) { return x > 0.f ? x : NEG_SLOPE * x; }

__device__ __forceinline__ void atomicMaxF(float* addr, float val) {
    unsigned int* ua = (unsigned int*)addr;
    unsigned int old = *ua;
    while (__uint_as_float(old) < val) {
        unsigned int assumed = old;
        old = atomicCAS(ua, assumed, __float_as_uint(val));
        if (old == assumed) break;
    }
}

// pass 1: segment max of leaky-relu logits over incoming edges
__global__ void edge_max_kernel(const int* __restrict__ src, const int* __restrict__ dst,
                                const float* __restrict__ el, const float* __restrict__ er,
                                float* __restrict__ nmax) {
    int i = blockIdx.x * blockDim.x + threadIdx.x;   // over E*H
    if (i >= EE * HH) return;
    const int e = i / HH, h = i % HH;
    const int s = src[e], d = dst[e];
    atomicMaxF(&nmax[d * HH + h], lrelu(el[s * HH + h] + er[d * HH + h]));
}

// pass 2: segment sum of exp(logit - max)
__global__ void edge_sum_kernel(const int* __restrict__ src, const int* __restrict__ dst,
                                const float* __restrict__ el, const float* __restrict__ er,
                                const float* __restrict__ nmax, float* __restrict__ nsum) {
    int i = blockIdx.x * blockDim.x + threadIdx.x;   // over E*H
    if (i >= EE * HH) return;
    const int e = i / HH, h = i % HH;
    const int s = src[e], d = dst[e];
    const float lg = lrelu(el[s * HH + h] + er[d * HH + h]);
    atomicAdd(&nsum[d * HH + h], __expf(lg - nmax[d * HH + h]));
}

// pass 3: alpha * feat[src] scatter-added into accum (per-etype sum fused here).
// One wave per (edge, head); each lane handles 2 contiguous dims.
__global__ void edge_msg_kernel(const int* __restrict__ src, const int* __restrict__ dst,
                                const float* __restrict__ el, const float* __restrict__ er,
                                const float* __restrict__ nmax, const float* __restrict__ nsum,
                                const float* __restrict__ feat, float* __restrict__ accum) {
    int wave = (blockIdx.x * (int)blockDim.x + (int)threadIdx.x) >> 5;
    if (wave >= EE * HH) return;
    const int lane = threadIdx.x & 31;
    const int e = wave >> 2;          // HH == 4
    const int h = wave & 3;
    const int s = src[e], d = dst[e];
    const float lg = lrelu(el[s * HH + h] + er[d * HH + h]);
    const float alpha = __expf(lg - nmax[d * HH + h]) / nsum[d * HH + h];
    const int dd = lane * 2;
    const float* fs = feat  + (size_t)s * HD + h * DD + dd;
    float*       od = accum + (size_t)d * HD + h * DD + dd;
    atomicAdd(od,     alpha * fs[0]);
    atomicAdd(od + 1, alpha * fs[1]);
}

// layer epilogue: h = maybe_relu(accum + sum_e bias[l,e]) emitted as bf16 for next GEMM
__global__ void finalize_layer(const float* __restrict__ accum,
                               const float* __restrict__ bias_l,  // [ETY][HD] at layer l
                               __bf16* __restrict__ hb, int do_relu) {
    long i = (long)blockIdx.x * blockDim.x + threadIdx.x;
    long stride = (long)gridDim.x * blockDim.x;
    const long total = (long)NN * HD;
    for (; i < total; i += stride) {
        const int c = (int)(i & (HD - 1));
        float v = accum[i] + bias_l[c] + bias_l[HD + c] + bias_l[2 * HD + c];
        if (do_relu) v = v > 0.f ? v : 0.f;
        hb[i] = (__bf16)v;
    }
}

// ---------------- host orchestration ----------------
extern "C" void kernel_launch(void* const* d_in, const int* in_sizes, int n_in,
                              void* d_out, int out_size, void* d_ws, size_t ws_size,
                              hipStream_t stream) {
    const float* x      = (const float*)d_in[0];
    const float* W      = (const float*)d_in[1];   // [L][ET][F][HD]
    const float* attn_l = (const float*)d_in[2];   // [L][ET][H][D]
    const float* attn_r = (const float*)d_in[3];
    const float* bias   = (const float*)d_in[4];   // [L][ET][HD]
    const float* W_out  = (const float*)d_in[5];   // [HD][C]
    const float* b_out  = (const float*)d_in[6];   // [C]
    const int*   src    = (const int*)d_in[7];     // [ET][E]
    const int*   dst    = (const int*)d_in[8];
    float* out = (float*)d_out;

    // workspace carve-out (256B aligned)
    char* ws = (char*)d_ws;
    size_t off = 0;
    auto carve = [&](size_t bytes) { char* p = ws + off; off = (off + bytes + 255) & ~(size_t)255; return p; };
    __bf16* hb     = (__bf16*)carve((size_t)NN * FF * sizeof(__bf16));     // 51.2 MB
    __bf16* Wb     = (__bf16*)carve((size_t)384 * FF * sizeof(__bf16));    // transposed weights (holds 349x256)
    float*  feat   = (float*)carve((size_t)NN * HD * sizeof(float));       // 102.4 MB
    float*  accum  = (float*)carve((size_t)NN * HD * sizeof(float));       // 102.4 MB
    float*  el     = (float*)carve((size_t)NN * HH * sizeof(float));
    float*  er     = (float*)carve((size_t)NN * HH * sizeof(float));
    float*  nmax   = (float*)carve((size_t)NN * HH * sizeof(float));
    float*  nsum   = (float*)carve((size_t)NN * HH * sizeof(float));

    const int TB = 256;
    const int fillBlocks = 2048;

    // h0 = x (bf16 copy for the WMMA path)
    f32_to_bf16<<<fillBlocks, TB, 0, stream>>>(x, hb, (long)NN * FF);

    for (int l = 0; l < LL; ++l) {
        fill_f32<<<fillBlocks, TB, 0, stream>>>(accum, 0.0f, (long)NN * HD);

        for (int e = 0; e < ETY; ++e) {
            const float* Wle = W + (((size_t)l * ETY + e) * FF * HD);
            const float* al  = attn_l + (((size_t)l * ETY + e) * HH * DD);
            const float* ar  = attn_r + (((size_t)l * ETY + e) * HH * DD);
            const int* src_e = src + (size_t)e * EE;
            const int* dst_e = dst + (size_t)e * EE;

            // W[l,e] -> bf16 transposed, then feat = hb @ W via WMMA
            f32_to_bf16_T<<<(FF * HD + TB - 1) / TB, TB, 0, stream>>>(Wle, Wb, FF, HD);
            {
                const int tiles = ((NN + 15) / 16) * (HD / 64);
                gemm_bf16_wmma<<<(tiles + 7) / 8, TB, 0, stream>>>(hb, Wb, nullptr, feat, NN, HD);
            }

            el_er_kernel<<<(NN * HH + TB - 1) / TB, TB, 0, stream>>>(feat, al, ar, el, er);

            fill_f32<<<(NN * HH + TB - 1) / TB, TB, 0, stream>>>(nmax, -1e30f, (long)NN * HH);
            fill_f32<<<(NN * HH + TB - 1) / TB, TB, 0, stream>>>(nsum, 0.0f, (long)NN * HH);

            edge_max_kernel<<<(EE * HH + TB - 1) / TB, TB, 0, stream>>>(src_e, dst_e, el, er, nmax);
            edge_sum_kernel<<<(EE * HH + TB - 1) / TB, TB, 0, stream>>>(src_e, dst_e, el, er, nmax, nsum);
            // one wave per (edge, head): E*H waves, 8 waves per block
            edge_msg_kernel<<<(EE * HH + 7) / 8, TB, 0, stream>>>(src_e, dst_e, el, er, nmax, nsum, feat, accum);
        }

        const float* bias_l = bias + (size_t)l * ETY * HD;
        finalize_layer<<<fillBlocks, TB, 0, stream>>>(accum, bias_l, hb, (l != LL - 1) ? 1 : 0);
    }

    // classifier: out = h @ W_out + b_out  (349 cols, guarded edge tiles)
    f32_to_bf16_T<<<(FF * CC + TB - 1) / TB, TB, 0, stream>>>(W_out, Wb, FF, CC);
    {
        const int tiles = ((NN + 15) / 16) * ((CC + 63) / 64);
        gemm_bf16_wmma<<<(tiles + 7) / 8, TB, 0, stream>>>(hb, Wb, b_out, out, NN, CC);
    }
}
